// ConstellationRelay_85031762526813
// MI455X (gfx1250) — matrix operand order repaired and checked by reading, not compiled, not run on hardware
//
#include <hip/hip_runtime.h>

// ---------------- types / helpers ----------------
typedef __attribute__((ext_vector_type(16))) __bf16       v16bf;
typedef __attribute__((ext_vector_type(8)))  float        v8f;
typedef __attribute__((ext_vector_type(8)))  unsigned int v8u;
typedef __attribute__((ext_vector_type(4)))  int          v4i;

typedef __attribute__((address_space(1))) v4i* gv4i_p;   // global (prints as __device__)
typedef __attribute__((address_space(3))) v4i* lv4i_p;   // LDS

union FragU { v8u u; v16bf b; };

__device__ __forceinline__ unsigned short f2bf(float f) {
  unsigned int u = __float_as_uint(f);
  unsigned int r = u + 0x7FFFu + ((u >> 16) & 1u);   // round-to-nearest-even
  return (unsigned short)(r >> 16);
}

__device__ __forceinline__ v8f wmma_bf16(v8u a, v8u b, v8f c) {
  FragU fa, fb; fa.u = a; fb.u = b;
  return __builtin_amdgcn_wmma_f32_16x16x32_bf16(
      false, fa.b, false, fb.b, (short)0, c, false, false);
}

// CDNA5 async global->LDS path (guarded: falls back to ds_store staging)
#if defined(__gfx1250__) && __has_builtin(__builtin_amdgcn_global_load_async_to_lds_b128) && \
    __has_builtin(__builtin_amdgcn_s_wait_asynccnt)
#define HAVE_ASYNC 1
#else
#define HAVE_ASYNC 0
#endif

// Problem constants
#define NTOK   32768   // B*S
#define DDIM   1024
#define NANCH  16
#define NCOMP  8
#define DC     64

// ---------------- prep kernels ----------------

// anchors (16 x 1024) -> l2-normalized bf16 rows
__global__ __launch_bounds__(256) void k_anchors(const float* __restrict__ anchors,
                                                 unsigned int* __restrict__ a_w) {
  __shared__ float red[256];
  const int row = blockIdx.x, tid = threadIdx.x;
  const float* ar = anchors + (long)row * DDIM;
  float s = 0.f;
  for (int e = tid; e < DDIM; e += 256) { float v = ar[e]; s += v * v; }
  red[tid] = s; __syncthreads();
  for (int off = 128; off > 0; off >>= 1) {
    if (tid < off) red[tid] += red[tid + off];
    __syncthreads();
  }
  float sc = 1.f / fmaxf(sqrtf(red[0]), 1e-12f);
  unsigned short* aw = (unsigned short*)a_w;
  for (int e = tid; e < DDIM; e += 256) aw[row * DDIM + e] = f2bf(ar[e] * sc);
}

// Pack Wp (512 x 1024 f32) into wave32 WMMA B-fragment layout:
// pb[kt(16)][nt(64)][lane(32)][v(8)] ; word = {K=kk+1, K=kk}
__global__ __launch_bounds__(256) void k_pack_wp(const float* __restrict__ Wp,
                                                 unsigned int* __restrict__ pb) {
  const int b = blockIdx.x;            // kt*64 + nt
  const int kt = b >> 6, nt = b & 63;
  const int tid = threadIdx.x;
  const int lane = tid >> 3, v = tid & 7;
  const int kk  = kt * 32 + ((lane >> 4) << 3) + ((v & 3) << 1) + ((v >> 2) << 4);
  const int col = nt * 16 + (lane & 15);
  unsigned int lo = f2bf(Wp[(long)kk * DDIM + col]);
  unsigned int hi = f2bf(Wp[(long)(kk + 1) * DDIM + col]);
  pb[(((long)b * 32 + lane) << 3) + v] = lo | (hi << 16);
}

// Pack W2 (8 x 128 x 64 f32) into B-fragment layout:
// pw2[comp(8)][kt(4)][nt(4)][lane(32)][v(8)]
__global__ __launch_bounds__(256) void k_pack_w2(const float* __restrict__ W2,
                                                 unsigned int* __restrict__ pw2) {
  const int b = blockIdx.x;            // comp*16 + kt*4 + nt
  const int comp = b >> 4, kt = (b >> 2) & 3, nt = b & 3;
  const int tid = threadIdx.x;
  const int lane = tid >> 3, v = tid & 7;
  const int kk  = kt * 32 + ((lane >> 4) << 3) + ((v & 3) << 1) + ((v >> 2) << 4);
  const int col = nt * 16 + (lane & 15);
  unsigned int lo = f2bf(W2[(comp * 128 + kk) * DC + col]);
  unsigned int hi = f2bf(W2[(comp * 128 + kk + 1) * DC + col]);
  pw2[(((long)b * 32 + lane) << 3) + v] = lo | (hi << 16);
}

// ---------------- kernel 1: per-16-token tile pipeline ----------------
// phases: LN+l2 -> tri WMMA -> layer1 sqReLU -> layer2 WMMA -> comp-LN -> packed y
__global__ __launch_bounds__(128) void k_tokens(
    const float* __restrict__ x, const float* __restrict__ ln_g, const float* __restrict__ ln_b,
    const unsigned int* __restrict__ a_w,
    const float* __restrict__ W1, const float* __restrict__ b1,
    const unsigned int* __restrict__ pw2,
    const float* __restrict__ b2, const float* __restrict__ cg, const float* __restrict__ cb,
    unsigned int* __restrict__ pa) {
  __shared__ unsigned int hbuf[16 * 512];   // 32KB: h tile bf16 pairs, reused as u tile
  __shared__ float part[4 * 16 * 16];       // 4KB: per-wave partial tri accumulators
  __shared__ float tri[16 * 16];            // 1KB
  __shared__ float ybuf[16 * 512];          // 32KB: pre-LN layer2 output (f32)

  const int tid  = threadIdx.x;
  const int lane = tid & 31;
  const int wv   = tid >> 5;
  const int tile = blockIdx.x;

  // -------- Phase 0: LayerNorm + L2-normalize, 4 tokens per wave --------
  unsigned short* hb = (unsigned short*)hbuf;
  for (int i = 0; i < 4; ++i) {
    const int lt = wv * 4 + i;
    const float* xr = x + ((long)tile * 16 + lt) * DDIM;
    float xv[32], s1 = 0.f, s2 = 0.f;
#pragma unroll
    for (int j = 0; j < 32; ++j) {
      float v = xr[j * 32 + lane];
      xv[j] = v; s1 += v; s2 += v * v;
    }
#pragma unroll
    for (int m = 16; m >= 1; m >>= 1) {
      s1 += __shfl_xor(s1, m, 32);
      s2 += __shfl_xor(s2, m, 32);
    }
    const float mu = s1 * (1.f / DDIM);
    const float rs = rsqrtf(s2 * (1.f / DDIM) - mu * mu + 1e-5f);
    float hv[32], q = 0.f;
#pragma unroll
    for (int j = 0; j < 32; ++j) {
      const int e = j * 32 + lane;
      float h = (xv[j] - mu) * rs * ln_g[e] + ln_b[e];
      hv[j] = h; q += h * h;
    }
#pragma unroll
    for (int m = 16; m >= 1; m >>= 1) q += __shfl_xor(q, m, 32);
    const float sc = 1.f / fmaxf(sqrtf(q), 1e-12f);
#pragma unroll
    for (int j = 0; j < 32; ++j) hb[lt * DDIM + j * 32 + lane] = f2bf(hv[j] * sc);
  }
  __syncthreads();

  // -------- Phase 1: tri = 1 - h @ a^T, K split over 4 waves --------
  {
    v8f acc = {};
    const int n16 = lane & 15;
    const int lhi = (lane >> 4) & 1;
    const int kb0 = wv * 256;
    for (int ks = 0; ks < 8; ++ks) {
      const int kb = kb0 + ks * 32;
      v8u av, bv;
#pragma unroll
      for (int v = 0; v < 8; ++v) {
        const int ka = kb + (lhi ? 8 : 0) + ((v & 3) << 1) + ((v >> 2) << 4); // A layout
        av[v] = hbuf[n16 * 512 + (ka >> 1)];
        const int kbk = kb + (lhi ? 8 : 0) + ((v & 3) << 1) + ((v >> 2) << 4); // B layout
        bv[v] = a_w[n16 * 512 + (kbk >> 1)];
      }
      acc = wmma_bf16(av, bv, acc);
    }
#pragma unroll
    for (int r = 0; r < 8; ++r) {
      const int m = r + (lhi << 3);
      part[(wv * 16 + m) * 16 + n16] = acc[r];
    }
  }
  __syncthreads();
  for (int idx = tid; idx < 256; idx += 128) {
    float s = part[idx] + part[256 + idx] + part[512 + idx] + part[768 + idx];
    tri[idx] = 1.f - s;
  }
  __syncthreads();

  // -------- Phase 2: layer1 (K=2) + squared ReLU -> bf16 u tile in LDS --------
  {
    const int lt = tid >> 3, comp = tid & 7;
    const float g0 = tri[lt * 16 + comp];
    const float g1 = tri[lt * 16 + 8 + comp];
    const float* w1a = W1 + (comp * 2 + 0) * 128;
    const float* w1b = W1 + (comp * 2 + 1) * 128;
    const float* b1r = b1 + comp * 128;
    unsigned short* ub = (unsigned short*)hbuf;
    for (int e = 0; e < 128; ++e) {
      float u = fmaf(g0, w1a[e], fmaf(g1, w1b[e], b1r[e]));
      u = fmaxf(u, 0.f); u *= u;
      ub[lt * DDIM + comp * 128 + e] = f2bf(u);
    }
  }
  __syncthreads();

  // -------- Phase 3: layer2 per compartment (M=16,K=128,N=64) via WMMA --------
  {
    const int n16 = lane & 15;
    const int lhi = (lane >> 4) & 1;
    for (int ci = 0; ci < 2; ++ci) {
      const int comp = wv * 2 + ci;
      v8f acc4[4] = {};
      for (int kt = 0; kt < 4; ++kt) {
        v8u av;
#pragma unroll
        for (int v = 0; v < 8; ++v) {
          const int ka = comp * 128 + kt * 32 + (lhi ? 8 : 0) + ((v & 3) << 1) + ((v >> 2) << 4);
          av[v] = hbuf[n16 * 512 + (ka >> 1)];
        }
#pragma unroll
        for (int nt = 0; nt < 4; ++nt) {
          const unsigned int* p = pw2 + ((((comp * 4 + kt) * 4 + nt) * 32 + lane) << 3);
          v8u bv;
#pragma unroll
          for (int v = 0; v < 8; ++v) bv[v] = p[v];
          acc4[nt] = wmma_bf16(av, bv, acc4[nt]);
        }
      }
#pragma unroll
      for (int nt = 0; nt < 4; ++nt)
#pragma unroll
        for (int r = 0; r < 8; ++r) {
          const int m = r + (lhi << 3);
          const int col = comp * 64 + nt * 16 + n16;
          ybuf[m * 512 + col] = acc4[nt][r] + b2[col];
        }
    }
  }
  __syncthreads();

  // -------- Phase 4: per-compartment LayerNorm (dc=64) -> packed A-fragment y --------
  {
    const int lt = tid >> 3, comp = tid & 7;
    const float* yr = ybuf + lt * 512 + comp * 64;
    float s1 = 0.f, s2 = 0.f;
    for (int e = 0; e < 64; ++e) { float v = yr[e]; s1 += v; s2 += v * v; }
    const float mu = s1 * (1.f / 64.f);
    const float rs = rsqrtf(s2 * (1.f / 64.f) - mu * mu + 1e-5f);
    const float* cgr = cg + comp * 64;
    const float* cbr = cb + comp * 64;
    unsigned short* pah = (unsigned short*)pa;
    for (int e = 0; e < 64; ++e) {
      const float v = (yr[e] - mu) * rs * cgr[e] + cbr[e];
      const int kg = comp * 64 + e;
      const int kt = kg >> 5, kin = kg & 31;
      const int ln2 = lt + ((kin & 8) ? 16 : 0);
      const int vv  = ((kin & 16) ? 4 : 0) + ((kin & 7) >> 1);
      const long ha = ((((long)tile * 16 + kt) * 32 + ln2) * 8 + vv) * 2 + (kin & 1);
      pah[ha] = f2bf(v);
    }
  }
}

// ---------------- kernel 2: update = y @ Wp ; out = x + sigmoid(gate)*(update+bp) ----
// Block = 8 waves. Each wave owns one 16-token m-tile; all waves share one 128-col
// B strip staged in LDS (async double-buffered global->LDS when available).
__global__ __launch_bounds__(256) void k_proj(
    const unsigned int* __restrict__ pa, const unsigned int* __restrict__ pb,
    const float* __restrict__ x, const float* __restrict__ bp,
    const float* __restrict__ gate, float* __restrict__ out) {
  __shared__ unsigned int lds_b[2][8 * 32 * 8];   // 2 x 8KB B-fragment buffers
  const int tid  = threadIdx.x, lane = tid & 31, wv = tid >> 5;
  const int mb    = blockIdx.x >> 3;              // 128-token group
  const int nbase = (blockIdx.x & 7) * 128;       // 128-col strip
  const int mtile = mb * 8 + wv;                  // this wave's 16-token tile
  const v8u* paV = (const v8u*)pa;

  v8f acc[8] = {};

  // cooperative stage of one k-step's B fragments (8KB contiguous) into LDS
  auto stage = [&](int kt, int buf) {
    const unsigned int* src =
        pb + (((long)kt * 64 + (nbase >> 4)) * 32) * 8 + (long)tid * 8;
#if HAVE_ASYNC
    __builtin_amdgcn_global_load_async_to_lds_b128(
        (gv4i_p)src, (lv4i_p)&lds_b[buf][tid * 8], 0, 0);
    __builtin_amdgcn_global_load_async_to_lds_b128(
        (gv4i_p)(src + 4), (lv4i_p)&lds_b[buf][tid * 8 + 4], 0, 0);
#else
    v8u t = *(const v8u*)src;
    *(v8u*)&lds_b[buf][tid * 8] = t;
#endif
  };

  stage(0, 0);
  for (int kt = 0; kt < 16; ++kt) {
    const int cur = kt & 1;
    if (kt < 15) stage(kt + 1, cur ^ 1);
#if HAVE_ASYNC
    // async loads complete in order: <=2 outstanding => current buffer is resident
    if (kt < 15) __builtin_amdgcn_s_wait_asynccnt(2);
    else         __builtin_amdgcn_s_wait_asynccnt(0);
#endif
    __syncthreads();                       // every wave has waited on its own portion
    const v8u av = paV[((long)mtile * 16 + kt) * 32 + lane];
#pragma unroll
    for (int nt = 0; nt < 8; ++nt) {
      const v8u bv = *(const v8u*)&lds_b[cur][(nt * 32 + lane) * 8];
      acc[nt] = wmma_bf16(av, bv, acc[nt]);
    }
    __syncthreads();                       // reads done before buffer is re-staged
  }

  const int n16 = lane & 15, lhi = lane >> 4;
#pragma unroll
  for (int nt = 0; nt < 8; ++nt) {
    const int col = nbase + nt * 16 + n16;
    const float sg  = 1.f / (1.f + __expf(-gate[col]));
    const float bpv = bp[col];
#pragma unroll
    for (int r = 0; r < 8; ++r) {
      const long idx = ((long)mtile * 16 + r + (lhi << 3)) * DDIM + col;
      out[idx] = x[idx] + sg * (acc[nt][r] + bpv);
    }
  }
}

// ---------------- launcher ----------------
extern "C" void kernel_launch(void* const* d_in, const int* in_sizes, int n_in,
                              void* d_out, int out_size, void* d_ws, size_t ws_size,
                              hipStream_t stream) {
  (void)in_sizes; (void)n_in; (void)out_size; (void)ws_size;
  const float* x       = (const float*)d_in[0];
  const float* anchors = (const float*)d_in[1];
  const float* ln_g    = (const float*)d_in[2];
  const float* ln_b    = (const float*)d_in[3];
  const float* W1      = (const float*)d_in[4];
  const float* b1      = (const float*)d_in[5];
  const float* W2      = (const float*)d_in[6];
  const float* b2      = (const float*)d_in[7];
  const float* cg      = (const float*)d_in[8];
  const float* cb      = (const float*)d_in[9];
  const float* Wp      = (const float*)d_in[10];
  const float* bp      = (const float*)d_in[11];
  const float* gate    = (const float*)d_in[12];
  float* out = (float*)d_out;

  unsigned int* ws  = (unsigned int*)d_ws;
  unsigned int* a_w = ws;                  //  16*512        =    8192 words
  unsigned int* pw2 = a_w + 8192;          //  8*4*4*32*8    =   32768 words
  unsigned int* pb  = pw2 + 32768;         //  16*64*32*8    =  262144 words
  unsigned int* pa  = pb + 262144;         //  2048*16*32*8  = 8388608 words (~34MB total)

  k_anchors<<<NANCH, 256, 0, stream>>>(anchors, a_w);
  k_pack_wp<<<16 * 64, 256, 0, stream>>>(Wp, pb);
  k_pack_w2<<<NCOMP * 16, 256, 0, stream>>>(W2, pw2);
  k_tokens<<<NTOK / 16, 128, 0, stream>>>(x, ln_g, ln_b, a_w, W1, b1, pw2, b2, cg, cb, pa);
  k_proj<<<(NTOK / 128) * (DDIM / 128), 256, 0, stream>>>(pa, pb, x, bp, gate, out);
}
